// RETAIN_62053687493243
// MI455X (gfx1250) — compile-verified, matrix operand order checked by reading, not run on hardware
//
#include <hip/hip_runtime.h>
#include <hip/hip_bf16.h>

// ---------------------------------------------------------------------------
// Problem constants (from the reference): B=16, T=128, DIN=1024, E=HA=HB=128.
// ---------------------------------------------------------------------------
constexpr int BB  = 16;    // batch
constexpr int T   = 128;   // sequence length
constexpr int DIN = 1024;  // input feature dim
constexpr int E   = 128;   // embedding dim
constexpr int H   = 128;   // GRU hidden dim (HA == HB)
constexpr int G   = 384;   // 3*H gate dim

typedef __bf16 bf16_t;
typedef __attribute__((ext_vector_type(16))) __bf16 v16bf;
typedef __attribute__((ext_vector_type(8)))  __bf16 v8bf;
typedef __attribute__((ext_vector_type(8)))  float  v8f;
typedef __attribute__((ext_vector_type(4)))  unsigned int v4u;
typedef __attribute__((ext_vector_type(8)))  int v8i;
typedef __attribute__((ext_vector_type(4)))  int v4i;

#if defined(__HIP_DEVICE_COMPILE__) && \
    __has_builtin(__builtin_amdgcn_tensor_load_to_lds) && \
    __has_builtin(__builtin_amdgcn_s_wait_tensorcnt)
#define USE_TDM 1
#else
#define USE_TDM 0
#endif

// Load a 16x32 bf16 A/B fragment half-row pair per the CDNA5 WMMA layout:
// lane holds row (lane&15); halves [0..7] = K chunk at kb, halves [8..15] =
// K chunk at kb+16 (kb = (lane>>4)*8 within the 32-wide K slice).
__device__ __forceinline__ v16bf frag2(const bf16_t* p0, const bf16_t* p1) {
  union U { v16bf v; v8bf h[2]; } u;
  u.h[0] = *reinterpret_cast<const v8bf*>(p0);
  u.h[1] = *reinterpret_cast<const v8bf*>(p1);
  return u.v;
}

__device__ __forceinline__ v8f wmma_bf16(v16bf a, v16bf b, v8f c) {
  // D(16x16,f32) = A(16x32,bf16) x B(32x16,bf16) + C
  return __builtin_amdgcn_wmma_f32_16x16x32_bf16(
      /*neg_a=*/false, a, /*neg_b=*/false, b,
      /*c_mod=*/(short)0, c, /*reuse_a=*/false, /*reuse_b=*/false);
}

__device__ __forceinline__ float sigmoidf_(float x) {
  return 1.0f / (1.0f + __expf(-x));
}

__device__ __forceinline__ unsigned lds_byte_off(const void* p) {
  // Generic LDS addresses map to the LDS allocation via addr[31:0]
  // (ISA 10.2 aperture table); the D# lds_addr takes that byte offset.
  return (unsigned)(unsigned long long)(uintptr_t)p;
}

// Issue one TDM flat 1-D copy: n8 x 8-byte elements global -> LDS.
// D# group0/group1 bitfields per CDNA5 ISA:
//   g0: count=1 | lds_addr | global_addr[56:0] | type=2
//   g1: data_size=3(8B) | tensor_dim0=n8 | tensor_dim1=1 | tile_dim0=n8 |
//       tile_dim1=1 | tensor_dim0_stride=n8
__device__ __forceinline__ void tdm_copy_to_lds(const void* gptr,
                                                unsigned lds_off, unsigned n8) {
#if USE_TDM
  unsigned long long ga = (unsigned long long)(uintptr_t)gptr;
  v4u g0;
  g0[0] = 1u;                              // count=1, user descriptor
  g0[1] = lds_off;                         // lds_addr (bytes)
  g0[2] = (unsigned)(ga & 0xFFFFFFFFu);    // global_addr[31:0] -> bits 95:64
  g0[3] = (unsigned)((ga >> 32) & 0x01FFFFFFu) | (2u << 30);  // addr[56:32]|type=2
  v8i g1;
  g1[0] = (int)(3u << 16);                 // workgroup_mask=0, data_size=3 (8B)
  g1[1] = (int)((n8 & 0xFFFFu) << 16);     // tensor_dim0[15:0] @ bits 63:48
  g1[2] = (int)(((n8 >> 16) & 0xFFFFu)     // tensor_dim0[31:16] @ bits 79:64
                | (1u << 16));             // tensor_dim1=1      @ bits 95:80
  g1[3] = (int)((n8 & 0xFFFFu) << 16);     // tile_dim0 @ bits 127:112
  g1[4] = 1;                               // tile_dim1=1, tile_dim2=0
  g1[5] = (int)n8;                         // tensor_dim0_stride[31:0]
  g1[6] = 0;                               // stride hi, dim1_stride lo
  g1[7] = 0;
  v4i z4 = {0, 0, 0, 0};
  v8i z8 = {0, 0, 0, 0, 0, 0, 0, 0};
  // amdgpu-toolchain (clang-23) 6-arg form: groups 0..3 + extra group + cpol.
  __builtin_amdgcn_tensor_load_to_lds(g0, g1, z4, z4, z8, 0);
#else
  (void)gptr; (void)lds_off; (void)n8;
#endif
}

__device__ __forceinline__ void coop_copy16(const void* g, void* l, int nbytes,
                                            int tid, int nthr) {
  const uint4* s = (const uint4*)g;
  uint4* d = (uint4*)l;
  for (int idx = tid; idx < nbytes / 16; idx += nthr) d[idx] = s[idx];
}

// ---------------------------------------------------------------------------
// fp32 -> bf16 cast
// ---------------------------------------------------------------------------
__global__ void cast_f32_to_bf16(const float* __restrict__ src,
                                 bf16_t* __restrict__ dst, int n) {
  int i = blockIdx.x * blockDim.x + threadIdx.x;
  if (i < n) dst[i] = (bf16_t)src[i];
}

// ---------------------------------------------------------------------------
// C(MxN,f32) = A(MxK,bf16) * B(NxK,bf16)^T + bias ; optional bf16 copy of C.
// One wave (32 threads) per 16x16 C tile.  K % 32 == 0, M,N % 16 == 0.
// ---------------------------------------------------------------------------
__global__ void __launch_bounds__(32) gemm_nt(
    const bf16_t* __restrict__ A, const bf16_t* __restrict__ Bw,
    const float* __restrict__ bias, float* __restrict__ C,
    bf16_t* __restrict__ Cb, int M, int N, int K) {
  const int mt = blockIdx.x, nt = blockIdx.y;
  const int lane = threadIdx.x;
  const int fm = lane & 15;        // row within A tile / row within B (=col of C)
  const int fg = lane >> 4;        // half-wave group selects K sub-chunk
  v8f acc = {};
  const bf16_t* Arow = A  + (size_t)(mt * 16 + fm) * K;
  const bf16_t* Brow = Bw + (size_t)(nt * 16 + fm) * K;
  for (int k0 = 0; k0 < K; k0 += 32) {
    const int kb = k0 + fg * 8;
    v16bf a = frag2(Arow + kb, Arow + kb + 16);
    v16bf b = frag2(Brow + kb, Brow + kb + 16);
    acc = wmma_bf16(a, b, acc);
  }
  const int n = nt * 16 + fm;
  const float bv = bias ? bias[n] : 0.0f;
#pragma unroll
  for (int v = 0; v < 8; ++v) {    // C layout: m = (lane>>4)*8 + v, n = lane&15
    const int m = mt * 16 + fg * 8 + v;
    const float val = acc[v] + bv;
    C[(size_t)m * N + n] = val;
    if (Cb) Cb[(size_t)m * N + n] = (bf16_t)val;
  }
}

// ---------------------------------------------------------------------------
// Fused dual-GRU scan + online softmax + context + output projection.
// One block per output position i (128 blocks), 128 threads = 4 waves.
// Loop-invariant weights (WhhA, WhhB, Wb: 229 KB bf16) are staged into LDS
// once via the Tensor Data Mover, so every sequential step's B-fragments come
// from ds_load instead of an L2 round-trip.  Total LDS ~284 KB (< 320 KB/WGP).
// ---------------------------------------------------------------------------
__global__ void __launch_bounds__(128) scan_fused(
    const float* __restrict__ giA, const float* __restrict__ giB,
    const float* __restrict__ emb,
    const bf16_t* __restrict__ WhhA, const bf16_t* __restrict__ WhhB,
    const bf16_t* __restrict__ Wbh,
    const float* __restrict__ bhhA, const float* __restrict__ bhhB,
    const float* __restrict__ Wa, const float* __restrict__ ba,
    const float* __restrict__ bb,
    const float* __restrict__ Wo, const float* __restrict__ bo,
    float* __restrict__ out) {
  __shared__ alignas(16) bf16_t sWhhA[G * H];   // 96 KB
  __shared__ alignas(16) bf16_t sWhhB[G * H];   // 96 KB
  __shared__ alignas(16) bf16_t sWb[E * H];     // 32 KB
  __shared__ alignas(16) bf16_t sHA[BB * H];    // GRU-A hidden state (bf16)
  __shared__ alignas(16) bf16_t sHB[BB * H];    // GRU-B hidden state (bf16)
  __shared__ alignas(16) bf16_t sHSb[BB * H];   // 0.5*hB (A-matrix for beta GEMM)
  __shared__ float sGH[BB * G];                 // gh scratch (16x384)
  __shared__ float sBeta[BB * E];               // tanh(hsB*Wb^T + bb)
  __shared__ float sAcc[BB * E];                // online-softmax context accum
  __shared__ float sPA[BB], sM[BB], sS[BB];     // pre_alpha, running max/sum
  __shared__ float sWa[H];

  const int i    = blockIdx.x;
  const int tid  = threadIdx.x;
  const int lane = tid & 31;
  const int wave = tid >> 5;
  const int fm   = lane & 15;
  const int fg   = lane >> 4;

  // --- stage loop-invariant weights into LDS (TDM if available) ---
#if USE_TDM
  if (tid < 32) {  // one wave issues the three DMA descriptors
    tdm_copy_to_lds(WhhA, lds_byte_off(sWhhA), (unsigned)(G * H * 2 / 8));
    tdm_copy_to_lds(WhhB, lds_byte_off(sWhhB), (unsigned)(G * H * 2 / 8));
    tdm_copy_to_lds(Wbh,  lds_byte_off(sWb),   (unsigned)(E * H * 2 / 8));
    __builtin_amdgcn_s_wait_tensorcnt(0);
  }
#else
  coop_copy16(WhhA, sWhhA, G * H * 2, tid, 128);
  coop_copy16(WhhB, sWhhB, G * H * 2, tid, 128);
  coop_copy16(Wbh,  sWb,   E * H * 2, tid, 128);
#endif

  // --- init scan state ---
  for (int idx = tid; idx < BB * H; idx += 128) {
    sHA[idx]  = (bf16_t)0.0f;
    sHB[idx]  = (bf16_t)0.0f;
    sHSb[idx] = (bf16_t)0.0f;
    sAcc[idx] = 0.0f;
  }
  if (tid < BB) { sM[tid] = -1e30f; sS[tid] = 0.0f; }
  sWa[tid] = Wa[tid];

  for (int k = 0; k <= i; ++k) {   // k>i terms have alpha==0: skip them
    const int t0 = i - k;
    __syncthreads();

    // ---- GRU-A: gh = hA(16x128) @ WhhA^T (384x128) ; 6 N-tiles per wave ----
    for (int tl = 0; tl < 6; ++tl) {
      const int nt = wave * 6 + tl;
      v8f c = {};
      const bf16_t* hrow = sHA + fm * H;
      const bf16_t* wrow = sWhhA + (nt * 16 + fm) * H;
#pragma unroll
      for (int kc = 0; kc < 4; ++kc) {
        const int kb = kc * 32 + fg * 8;
        c = wmma_bf16(frag2(hrow + kb, hrow + kb + 16),
                      frag2(wrow + kb, wrow + kb + 16), c);
      }
#pragma unroll
      for (int v = 0; v < 8; ++v)
        sGH[(fg * 8 + v) * G + nt * 16 + fm] = c[v];
    }
    __syncthreads();

    // ---- GRU-A gate math: thread owns row b = tid>>3, 16 h-elements ----
    {
      const int b = tid >> 3, e0 = (tid & 7) * 16;
      const float* gi = giA + (size_t)(b * T + t0) * G;
#pragma unroll
      for (int j = 0; j < 16; ++j) {
        const int hh = e0 + j;
        const float gr = gi[hh]         + sGH[b * G + hh]         + bhhA[hh];
        const float gz = gi[H + hh]     + sGH[b * G + H + hh]     + bhhA[H + hh];
        const float hn =                  sGH[b * G + 2 * H + hh] + bhhA[2 * H + hh];
        const float gn = gi[2 * H + hh];
        const float r  = sigmoidf_(gr);
        const float z  = sigmoidf_(gz);
        const float nv = tanhf(gn + r * hn);
        const float hp = (float)sHA[b * H + hh];
        const float h2 = (1.0f - z) * nv + z * hp;
        sHA[b * H + hh] = (bf16_t)h2;
      }
    }
    __syncthreads();

    // ---- pre_alpha = dot(0.5*hA, Wa) + ba (16 rows) ----
    if (tid < BB) {
      float s = 0.0f;
      for (int e = 0; e < H; ++e) s += 0.5f * (float)sHA[tid * H + e] * sWa[e];
      sPA[tid] = s + ba[0];
    }

    // ---- GRU-B: gh = hB @ WhhB^T ----
    for (int tl = 0; tl < 6; ++tl) {
      const int nt = wave * 6 + tl;
      v8f c = {};
      const bf16_t* hrow = sHB + fm * H;
      const bf16_t* wrow = sWhhB + (nt * 16 + fm) * H;
#pragma unroll
      for (int kc = 0; kc < 4; ++kc) {
        const int kb = kc * 32 + fg * 8;
        c = wmma_bf16(frag2(hrow + kb, hrow + kb + 16),
                      frag2(wrow + kb, wrow + kb + 16), c);
      }
#pragma unroll
      for (int v = 0; v < 8; ++v)
        sGH[(fg * 8 + v) * G + nt * 16 + fm] = c[v];
    }
    __syncthreads();

    // ---- GRU-B gate math; also emit 0.5*h as bf16 for the beta GEMM ----
    {
      const int b = tid >> 3, e0 = (tid & 7) * 16;
      const float* gi = giB + (size_t)(b * T + t0) * G;
#pragma unroll
      for (int j = 0; j < 16; ++j) {
        const int hh = e0 + j;
        const float gr = gi[hh]         + sGH[b * G + hh]         + bhhB[hh];
        const float gz = gi[H + hh]     + sGH[b * G + H + hh]     + bhhB[H + hh];
        const float hn =                  sGH[b * G + 2 * H + hh] + bhhB[2 * H + hh];
        const float gn = gi[2 * H + hh];
        const float r  = sigmoidf_(gr);
        const float z  = sigmoidf_(gz);
        const float nv = tanhf(gn + r * hn);
        const float hp = (float)sHB[b * H + hh];
        const float h2 = (1.0f - z) * nv + z * hp;
        sHB[b * H + hh]  = (bf16_t)h2;
        sHSb[b * H + hh] = (bf16_t)(0.5f * h2);
      }
    }
    __syncthreads();

    // ---- beta = tanh(hsB(16x128) @ Wb^T (128x128) + bb); 2 N-tiles/wave ----
    for (int tl = 0; tl < 2; ++tl) {
      const int nt = wave * 2 + tl;
      v8f c = {};
      const bf16_t* hrow = sHSb + fm * H;
      const bf16_t* wrow = sWb + (nt * 16 + fm) * H;
#pragma unroll
      for (int kc = 0; kc < 4; ++kc) {
        const int kb = kc * 32 + fg * 8;
        c = wmma_bf16(frag2(hrow + kb, hrow + kb + 16),
                      frag2(wrow + kb, wrow + kb + 16), c);
      }
#pragma unroll
      for (int v = 0; v < 8; ++v) {
        const int col = nt * 16 + fm;
        sBeta[(fg * 8 + v) * E + col] = tanhf(c[v] + bb[col]);
      }
    }
    __syncthreads();

    // ---- online softmax over k: acc = acc*sc + p * beta * emb[b, i-k] ----
    {
      const int b = tid >> 3, e0 = (tid & 7) * 16;
      const float pa  = sPA[b];
      const float mo  = sM[b];
      const float mn2 = fmaxf(mo, pa);
      const float sc  = __expf(mo - mn2);
      const float p   = __expf(pa - mn2);
      const float* er = emb + (size_t)(b * T + t0) * E;
#pragma unroll
      for (int j = 0; j < 16; ++j) {
        const int e = e0 + j;
        sAcc[b * E + e] = sAcc[b * E + e] * sc + p * sBeta[b * E + e] * er[e];
      }
      if ((tid & 7) == 0) { sS[b] = sS[b] * sc + p; sM[b] = mn2; }
    }
  }

  __syncthreads();
  // ---- logits[b, i] = dot(acc/s, Wo) + bo ----
  if (tid < BB) {
    const float inv = 1.0f / sS[tid];
    float s = 0.0f;
    for (int e = 0; e < E; ++e) s += sAcc[tid * E + e] * Wo[e];
    out[tid * T + i] = s * inv + bo[0];
  }
}

// ---------------------------------------------------------------------------
// Launch
// ---------------------------------------------------------------------------
extern "C" void kernel_launch(void* const* d_in, const int* in_sizes, int n_in,
                              void* d_out, int out_size, void* d_ws, size_t ws_size,
                              hipStream_t stream) {
  (void)in_sizes; (void)n_in; (void)out_size; (void)ws_size;
  const float* x     = (const float*)d_in[0];
  const float* W_emb = (const float*)d_in[1];
  const float* Wih_a = (const float*)d_in[2];
  const float* Whh_a = (const float*)d_in[3];
  const float* bih_a = (const float*)d_in[4];
  const float* bhh_a = (const float*)d_in[5];
  const float* Wa    = (const float*)d_in[6];
  const float* ba    = (const float*)d_in[7];
  const float* Wih_b = (const float*)d_in[8];
  const float* Whh_b = (const float*)d_in[9];
  const float* bih_b = (const float*)d_in[10];
  const float* bhh_b = (const float*)d_in[11];
  const float* Wb    = (const float*)d_in[12];
  const float* bb    = (const float*)d_in[13];
  const float* Wo    = (const float*)d_in[14];
  const float* bo    = (const float*)d_in[15];
  float* out = (float*)d_out;

  // Workspace bump allocator (256B aligned). Total ~12.6 MB.
  size_t off = 0;
  auto alloc = [&](size_t bytes) -> char* {
    char* p = (char*)d_ws + off;
    off = (off + bytes + 255) & ~(size_t)255;
    return p;
  };
  bf16_t* xh     = (bf16_t*)alloc((size_t)BB * T * DIN * 2);
  bf16_t* W_embh = (bf16_t*)alloc((size_t)E * DIN * 2);
  bf16_t* WihA_h = (bf16_t*)alloc((size_t)G * E * 2);
  bf16_t* WhhA_h = (bf16_t*)alloc((size_t)G * H * 2);
  bf16_t* WihB_h = (bf16_t*)alloc((size_t)G * E * 2);
  bf16_t* WhhB_h = (bf16_t*)alloc((size_t)G * H * 2);
  bf16_t* Wb_h   = (bf16_t*)alloc((size_t)E * H * 2);
  float*  emb    = (float*)alloc((size_t)BB * T * E * 4);
  bf16_t* embh   = (bf16_t*)alloc((size_t)BB * T * E * 2);
  float*  giA    = (float*)alloc((size_t)BB * T * G * 4);
  float*  giB    = (float*)alloc((size_t)BB * T * G * 4);

  auto cast = [&](const float* s, bf16_t* d, int n) {
    cast_f32_to_bf16<<<(n + 255) / 256, 256, 0, stream>>>(s, d, n);
  };
  cast(x,     xh,     BB * T * DIN);
  cast(W_emb, W_embh, E * DIN);
  cast(Wih_a, WihA_h, G * E);
  cast(Whh_a, WhhA_h, G * H);
  cast(Wih_b, WihB_h, G * E);
  cast(Whh_b, WhhB_h, G * H);
  cast(Wb,    Wb_h,   E * H);

  // emb = x @ W_emb^T  (2048 x 128 x K=1024), + bf16 copy for downstream GEMMs
  gemm_nt<<<dim3((BB * T) / 16, E / 16), 32, 0, stream>>>(
      xh, W_embh, nullptr, emb, embh, BB * T, E, DIN);
  // Hoisted input-gate GEMMs: gi = emb @ Wih^T + bih  (2048 x 384 x K=128)
  gemm_nt<<<dim3((BB * T) / 16, G / 16), 32, 0, stream>>>(
      embh, WihA_h, bih_a, giA, nullptr, BB * T, G, E);
  gemm_nt<<<dim3((BB * T) / 16, G / 16), 32, 0, stream>>>(
      embh, WihB_h, bih_b, giB, nullptr, BB * T, G, E);

  // Fused recurrence + attention + output projection: one block per i.
  scan_fused<<<T, 128, 0, stream>>>(giA, giB, emb, WhhA_h, WhhB_h, Wb_h,
                                    bhh_a, bhh_b, Wa, ba, bb, Wo, bo, out);
}
